// SpatialTemporalAttention_87694642250281
// MI455X (gfx1250) — matrix operand order repaired and checked by reading, not compiled, not run on hardware
//
#include <hip/hip_runtime.h>
#include <hip/hip_bf16.h>

typedef __attribute__((ext_vector_type(16))) __bf16        v16bf;
typedef __attribute__((ext_vector_type(8)))  float         v8f;
typedef __attribute__((ext_vector_type(4)))  unsigned int  u32x4v;
typedef __attribute__((ext_vector_type(4)))  int           i32x4v;
typedef __attribute__((ext_vector_type(8)))  int           i32x8v;

union FragBF {
    v16bf v;
    ulonglong2 q[2];
};

#if __has_builtin(__builtin_amdgcn_tensor_load_to_lds)
#define ATTN_HAS_TDM 1
#else
#define ATTN_HAS_TDM 0
#endif

// ---------------------------------------------------------------------------
// WMMA helpers (gfx1250: V_WMMA_F32_16X16X32_BF16, D = A*B + C)
// A-frag: lane L holds row (L&15); 16 contiguous K
//         starting at (L<16 ? 0 : 16).
// B-frag: lane L holds column (L&15) of B (= row of B^T); 16 contiguous K
//         starting at (L<16 ? 0 : 16).
// C/D   : VGPR r holds M = r + (L<16?0:8), N = L&15.
// ---------------------------------------------------------------------------
__device__ __forceinline__ v8f wmma_bf16(v16bf a, v16bf b, v8f c) {
    return __builtin_amdgcn_wmma_f32_16x16x32_bf16(
        false, a, false, b, (short)0, c, false, false);
}

__device__ __forceinline__ v16bf load_a_frag(const __bf16* base, int ld) {
    const int lane = threadIdx.x & 31;
    const int row  = lane & 15;
    const int koff = (lane < 16) ? 0 : 16;
    const __bf16* p = base + (size_t)row * ld + koff;
    FragBF f;
    f.q[0] = *reinterpret_cast<const ulonglong2*>(p);
    f.q[1] = *reinterpret_cast<const ulonglong2*>(p + 8);
    return f.v;
}

__device__ __forceinline__ v16bf load_b_frag(const __bf16* base, int ld) {
    const int lane = threadIdx.x & 31;
    const int col  = lane & 15;
    const int koff = (lane < 16) ? 0 : 16;
    const __bf16* p = base + (size_t)col * ld + koff;
    FragBF f;
    f.q[0] = *reinterpret_cast<const ulonglong2*>(p);
    f.q[1] = *reinterpret_cast<const ulonglong2*>(p + 8);
    return f.v;
}

// ---------------------------------------------------------------------------
// Tensor Data Mover: DMA a [64 rows x 32 K] bf16 tile (row stride = ldb
// elements) from global into LDS at byte offset lds_off.
// D# group0: count=1 | lds_addr | global_addr(57b) | type=2.
// D# group1: data_size=2B, tensor_dim0=ldb, tensor_dim1=nrows,
//            tile_dim0=32, tile_dim1=64, tensor_dim0_stride=ldb.
// Groups 2/3 zero (2-D tile).
// ---------------------------------------------------------------------------
#if ATTN_HAS_TDM
__device__ __forceinline__ void tdm_load_tile(unsigned lds_off,
                                              const __bf16* gptr,
                                              int ldb, int nrows) {
    unsigned long long ga = (unsigned long long)(size_t)gptr;
    u32x4v g0;
    g0[0] = 1u;                                     // count=1, user descriptor
    g0[1] = lds_off;                                // lds_addr (bytes)
    g0[2] = (unsigned)(ga & 0xffffffffu);           // global_addr[31:0]
    g0[3] = (unsigned)((ga >> 32) & 0x1ffffffu)     // global_addr[56:32]
          | (2u << 30);                             // type=2 ("image")
    i32x8v g1;
    g1[0] = (1 << 16);                              // data_size=1 -> 2 bytes
    g1[1] = (int)((ldb & 0xffff) << 16);            // tensor_dim0 lo16
    g1[2] = (int)(((unsigned)ldb >> 16) & 0xffff)   // tensor_dim0 hi16
          | (int)((nrows & 0xffff) << 16);          // tensor_dim1 lo16
    g1[3] = (int)(((unsigned)nrows >> 16) & 0xffff) // tensor_dim1 hi16
          | (32 << 16);                             // tile_dim0 = 32
    g1[4] = 64;                                     // tile_dim1 = 64 (dim2=0)
    g1[5] = ldb;                                    // tensor_dim0_stride lo32
    g1[6] = 0;
    g1[7] = 0;
    i32x4v z4 = {0, 0, 0, 0};
#if __clang_major__ >= 23
    i32x8v z8 = {0, 0, 0, 0, 0, 0, 0, 0};
    __builtin_amdgcn_tensor_load_to_lds(g0, g1, z4, z4, z8, 0);
#else
    __builtin_amdgcn_tensor_load_to_lds(g0, g1, z4, z4, 0);
#endif
}

__device__ __forceinline__ void wait_tensorcnt_le1() {
#if __has_builtin(__builtin_amdgcn_s_wait_tensorcnt)
    __builtin_amdgcn_s_wait_tensorcnt((short)1);
#else
    asm volatile("s_wait_tensorcnt 1" ::: "memory");
#endif
}
__device__ __forceinline__ void wait_tensorcnt_le0() {
#if __has_builtin(__builtin_amdgcn_s_wait_tensorcnt)
    __builtin_amdgcn_s_wait_tensorcnt((short)0);
#else
    asm volatile("s_wait_tensorcnt 0" ::: "memory");
#endif
}
#endif  // ATTN_HAS_TDM

// ---------------------------------------------------------------------------
// Block-cooperative 64x64 GEMM mainloop. 4 waves; wave w owns rows
// [w*16, w*16+16) of the 64-row tile; all waves share the B tile staged in
// LDS (double-buffered, DMA'd by the Tensor Data Mover from wave 0).
// Awave: this wave's 16-row A block at k=0 (row-major, lda).
// Btile: 64-row B^T block at k=0 (row-major, ldb).
// ---------------------------------------------------------------------------
__device__ __forceinline__ void gemm_tile64(const __bf16* __restrict__ Awave,
                                            int lda,
                                            const __bf16* __restrict__ Btile,
                                            int ldb, int nrows, int K,
                                            __bf16* lds, v8f acc[4]) {
    const int wslot = threadIdx.x >> 5;
#if ATTN_HAS_TDM
    const unsigned lds_off = (unsigned)(size_t)lds;
    int cur = 0;
    if (wslot == 0) tdm_load_tile(lds_off, Btile, ldb, nrows);
    for (int kk = 0; kk < K; kk += 32) {
        const bool more = (kk + 32) < K;
        if (wslot == 0) {
            if (more) {
                tdm_load_tile(lds_off + (unsigned)((cur ^ 1) * 64 * 32 * 2),
                              Btile + kk + 32, ldb, nrows);
                wait_tensorcnt_le1();   // retire DMA for buf[cur]
            } else {
                wait_tensorcnt_le0();
            }
        }
        __syncthreads();                // publish buf[cur] to all waves
        __builtin_prefetch(Awave + kk + 64, 0, 1);
        v16bf a = load_a_frag(Awave + kk, lda);
        const __bf16* bbase = lds + cur * (64 * 32);
#pragma unroll
        for (int t = 0; t < 4; ++t) {
            v16bf bb = load_b_frag(bbase + (size_t)(t * 16) * 32, 32);
            acc[t] = wmma_bf16(a, bb, acc[t]);
        }
        __syncthreads();                // buf[cur] free for reuse (dist 2)
        cur ^= 1;
    }
#else
    // Fallback: cooperative staging, single buffer.
    const int tid = threadIdx.x;
    for (int kk = 0; kk < K; kk += 32) {
        const __bf16* src = Btile + (size_t)(tid >> 1) * ldb + kk + (tid & 1) * 16;
        __bf16* dst = lds + (size_t)(tid >> 1) * 32 + (tid & 1) * 16;
        *reinterpret_cast<ulonglong2*>(dst)     = *reinterpret_cast<const ulonglong2*>(src);
        *reinterpret_cast<ulonglong2*>(dst + 8) = *reinterpret_cast<const ulonglong2*>(src + 8);
        __syncthreads();
        v16bf a = load_a_frag(Awave + kk, lda);
#pragma unroll
        for (int t = 0; t < 4; ++t) {
            v16bf bb = load_b_frag(lds + (size_t)(t * 16) * 32, 32);
            acc[t] = wmma_bf16(a, bb, acc[t]);
        }
        __syncthreads();
    }
    (void)nrows;
#endif
}

// ---------------------------------------------------------------------------
// Elementwise f32 -> bf16 conversion (4 elems/thread)
// ---------------------------------------------------------------------------
__global__ void convert_f32_bf16(const float* __restrict__ src,
                                 __bf16* __restrict__ dst, int count) {
    int i = (blockIdx.x * blockDim.x + threadIdx.x) * 4;
    if (i + 3 < count) {
        float4 v = *reinterpret_cast<const float4*>(src + i);
        union { __bf16 h[4]; unsigned long long u; } o;
        o.h[0] = (__bf16)v.x; o.h[1] = (__bf16)v.y;
        o.h[2] = (__bf16)v.z; o.h[3] = (__bf16)v.w;
        *reinterpret_cast<unsigned long long*>(dst + i) = o.u;
    }
}

// ---------------------------------------------------------------------------
// Fused QKV projection: [2048x1024] x [1536x1024]^T per batch.
// Block tile 64x64 (TDM-staged weights). Q,K stored [b,n,512];
// V stored transposed vT[b,h,64,2048].
// ---------------------------------------------------------------------------
__global__ void qkv_gemm(const __bf16* __restrict__ X,
                         const __bf16* __restrict__ Wqkv,
                         __bf16* __restrict__ Q, __bf16* __restrict__ Kb,
                         __bf16* __restrict__ VT) {
    __shared__ __align__(16) __bf16 ldsB[2 * 64 * 32];
    const int lane  = threadIdx.x & 31;
    const int wslot = threadIdx.x >> 5;
    const int blk   = blockIdx.x;
    const int b     = blk / 768;
    const int rem   = blk % 768;
    const int ti    = rem / 24, tj = rem % 24;

    const __bf16* Awave = X + ((size_t)b * 2048 + ti * 64 + wslot * 16) * 1024;
    const __bf16* Btile = Wqkv + (size_t)(tj * 64) * 1024;
    v8f acc[4] = {};
    gemm_tile64(Awave, 1024, Btile, 1024, 1536, 1024, ldsB, acc);

    const int half = (lane < 16) ? 0 : 8;
    const int col  = lane & 15;
#pragma unroll
    for (int t = 0; t < 4; ++t) {
        int n = tj * 64 + t * 16 + col;
#pragma unroll
        for (int r = 0; r < 8; ++r) {
            int i = ti * 64 + wslot * 16 + r + half;
            __bf16 bv = (__bf16)acc[t][r];
            if (n < 512) {
                Q[((size_t)b * 2048 + i) * 512 + n] = bv;
            } else if (n < 1024) {
                Kb[((size_t)b * 2048 + i) * 512 + (n - 512)] = bv;
            } else {
                int c = n - 1024, hh = c >> 6, dd = c & 63;
                VT[(((size_t)b * 8 + hh) * 64 + dd) * 2048 + i] = bv;
            }
        }
    }
}

// ---------------------------------------------------------------------------
// Mask GEMM: sim = x . x^T (sign-invariant to normalization); mask = sim > 0.
// ---------------------------------------------------------------------------
__global__ void mask_gemm(const __bf16* __restrict__ X,
                          unsigned char* __restrict__ Mask) {
    __shared__ __align__(16) __bf16 ldsB[2 * 64 * 32];
    const int lane  = threadIdx.x & 31;
    const int wslot = threadIdx.x >> 5;
    const int blk   = blockIdx.x;
    const int b     = blk >> 10;
    const int rem   = blk & 1023;
    const int ti    = rem >> 5, tj = rem & 31;

    const __bf16* Awave = X + ((size_t)b * 2048 + ti * 64 + wslot * 16) * 1024;
    const __bf16* Btile = X + ((size_t)b * 2048 + tj * 64) * 1024;
    v8f acc[4] = {};
    gemm_tile64(Awave, 1024, Btile, 1024, 2048, 1024, ldsB, acc);

    const int half = (lane < 16) ? 0 : 8;
    const int col  = lane & 15;
    unsigned char* mb = Mask + (size_t)b * 2048 * 2048;
#pragma unroll
    for (int t = 0; t < 4; ++t) {
        int j = tj * 64 + t * 16 + col;
#pragma unroll
        for (int r = 0; r < 8; ++r) {
            int i = ti * 64 + wslot * 16 + r + half;
            mb[(size_t)i * 2048 + j] = (acc[t][r] > 0.0f) ? 1 : 0;
        }
    }
}

// ---------------------------------------------------------------------------
// Flash attention per (b, h, 16-row query tile). Online softmax; additive
// -1e9 bias where mask==0 (matches reference: fully-biased rows reduce to a
// plain softmax). P transposed C-layout -> A-layout through per-wave LDS.
// ---------------------------------------------------------------------------
__global__ void attn_kernel(const __bf16* __restrict__ Q,
                            const __bf16* __restrict__ Kb,
                            const __bf16* __restrict__ VT,
                            const unsigned char* __restrict__ Mask,
                            const int* __restrict__ enhance,
                            __bf16* __restrict__ O) {
    __shared__ __align__(16) __bf16 ldsP[4][16 * 32];
    const int wslot = threadIdx.x >> 5;
    const int wid   = blockIdx.x * 4 + wslot;
    const int lane  = threadIdx.x & 31;
    const int b     = wid >> 10;
    const int rem   = wid & 1023;
    const int h     = rem >> 7;
    const int i0    = (rem & 127) << 4;
    const int half  = (lane < 16) ? 0 : 8;
    const int col   = lane & 15;
    const int enh   = *enhance;

    const __bf16* qbase = Q + ((size_t)b * 2048 + i0) * 512 + h * 64;
    v16bf aq0 = load_a_frag(qbase, 512);
    v16bf aq1 = load_a_frag(qbase + 32, 512);

    float m[8], l[8];
#pragma unroll
    for (int r = 0; r < 8; ++r) { m[r] = -1e30f; l[r] = 0.0f; }
    v8f accO[4] = {};

    const unsigned char* mrow = Mask + (size_t)b * 2048 * 2048;
    __bf16* pbuf = ldsP[wslot];

    for (int j0 = 0; j0 < 2048; j0 += 32) {
        v8f s[2] = {};
#pragma unroll
        for (int u = 0; u < 2; ++u) {
            const __bf16* kbase = Kb + ((size_t)b * 2048 + j0 + u * 16) * 512 + h * 64;
            v16bf bk0 = load_b_frag(kbase, 512);
            v16bf bk1 = load_b_frag(kbase + 32, 512);
            s[u] = wmma_bf16(aq0, bk0, s[u]);
            s[u] = wmma_bf16(aq1, bk1, s[u]);
        }
#pragma unroll
        for (int u = 0; u < 2; ++u) {
#pragma unroll
            for (int r = 0; r < 8; ++r) {
                int i = i0 + r + half;
                int j = j0 + u * 16 + col;
                unsigned char mb = enh ? mrow[(size_t)i * 2048 + j] : (unsigned char)1;
                s[u][r] = s[u][r] * 0.125f + (mb ? 0.0f : -1.0e9f);
            }
        }
#pragma unroll
        for (int r = 0; r < 8; ++r) {
            float rmax = fmaxf(s[0][r], s[1][r]);
#pragma unroll
            for (int d = 1; d < 16; d <<= 1) rmax = fmaxf(rmax, __shfl_xor(rmax, d, 32));
            float nm   = fmaxf(m[r], rmax);
            float corr = __expf(m[r] - nm);
            float p0   = __expf(s[0][r] - nm);
            float p1   = __expf(s[1][r] - nm);
            float rs   = p0 + p1;
#pragma unroll
            for (int d = 1; d < 16; d <<= 1) rs += __shfl_xor(rs, d, 32);
            l[r] = l[r] * corr + rs;
            m[r] = nm;
#pragma unroll
            for (int t = 0; t < 4; ++t) accO[t][r] *= corr;
            int row = r + half;
            pbuf[row * 32 + col]      = (__bf16)p0;
            pbuf[row * 32 + 16 + col] = (__bf16)p1;
        }
        asm volatile("s_wait_dscnt 0" ::: "memory");  // own-wave LDS RAW
        v16bf ap = load_a_frag(pbuf, 32);
#pragma unroll
        for (int t = 0; t < 4; ++t) {
            v16bf vb = load_b_frag(VT + (((size_t)b * 8 + h) * 64 + t * 16) * 2048 + j0, 2048);
            accO[t] = wmma_bf16(ap, vb, accO[t]);
        }
    }
#pragma unroll
    for (int t = 0; t < 4; ++t) {
#pragma unroll
        for (int r = 0; r < 8; ++r) {
            int i = i0 + r + half;
            int c = h * 64 + t * 16 + col;
            O[((size_t)b * 2048 + i) * 512 + c] = (__bf16)(accO[t][r] / l[r]);
        }
    }
}

// ---------------------------------------------------------------------------
// Output projection: Y = O @ Wo^T + bo  (f32 result), TDM-staged weights.
// ---------------------------------------------------------------------------
__global__ void out_gemm(const __bf16* __restrict__ O,
                         const __bf16* __restrict__ Wob,
                         const float* __restrict__ bo,
                         float* __restrict__ Y) {
    __shared__ __align__(16) __bf16 ldsB[2 * 64 * 32];
    const int lane  = threadIdx.x & 31;
    const int wslot = threadIdx.x >> 5;
    const int blk   = blockIdx.x;
    const int b     = blk >> 9;
    const int rem   = blk & 511;
    const int ti    = rem >> 4, tj = rem & 15;

    const __bf16* Awave = O + ((size_t)b * 2048 + ti * 64 + wslot * 16) * 512;
    const __bf16* Btile = Wob + (size_t)(tj * 64) * 512;
    v8f acc[4] = {};
    gemm_tile64(Awave, 512, Btile, 512, 1024, 512, ldsB, acc);

    const int half = (lane < 16) ? 0 : 8;
    const int col  = lane & 15;
#pragma unroll
    for (int t = 0; t < 4; ++t) {
        int n = tj * 64 + t * 16 + col;
        float bias = bo[n];
#pragma unroll
        for (int r = 0; r < 8; ++r) {
            int i = ti * 64 + wslot * 16 + r + half;
            Y[((size_t)b * 2048 + i) * 1024 + n] = acc[t][r] + bias;
        }
    }
}

// ---------------------------------------------------------------------------
extern "C" void kernel_launch(void* const* d_in, const int* in_sizes, int n_in,
                              void* d_out, int out_size, void* d_ws, size_t ws_size,
                              hipStream_t stream) {
    const float* x  = (const float*)d_in[0];   // [4,2048,1024]
    const float* Wq = (const float*)d_in[1];   // [512,1024]
    const float* Wk = (const float*)d_in[2];
    const float* Wv = (const float*)d_in[3];
    const float* Wo = (const float*)d_in[4];   // [1024,512]
    const float* bo = (const float*)d_in[5];   // [1024]
    const int* enhance = (const int*)d_in[6];
    float* out = (float*)d_out;

    char* ws = (char*)d_ws;
    size_t off = 0;
    auto carve = [&](size_t bytes) {
        void* p = ws + off;
        off += (bytes + 255) & ~(size_t)255;
        return p;
    };
    __bf16* xbf  = (__bf16*)carve((size_t)4 * 2048 * 1024 * 2);   // 16 MB
    __bf16* wqkv = (__bf16*)carve((size_t)1536 * 1024 * 2);       //  3 MB
    __bf16* wob  = (__bf16*)carve((size_t)1024 * 512 * 2);        //  1 MB
    __bf16* qb   = (__bf16*)carve((size_t)4 * 2048 * 512 * 2);    //  8 MB
    __bf16* kb   = (__bf16*)carve((size_t)4 * 2048 * 512 * 2);    //  8 MB
    __bf16* vt   = (__bf16*)carve((size_t)4 * 8 * 64 * 2048 * 2); //  8 MB
    __bf16* ob   = (__bf16*)carve((size_t)4 * 2048 * 512 * 2);    //  8 MB
    unsigned char* mask = (unsigned char*)carve((size_t)4 * 2048 * 2048); // 16 MB
    (void)ws_size; (void)in_sizes; (void)n_in; (void)out_size;

    {
        int cx = 4 * 2048 * 1024;
        convert_f32_bf16<<<cx / (256 * 4), 256, 0, stream>>>(x, xbf, cx);
        int cw = 512 * 1024;
        convert_f32_bf16<<<cw / (256 * 4), 256, 0, stream>>>(Wq, wqkv, cw);
        convert_f32_bf16<<<cw / (256 * 4), 256, 0, stream>>>(Wk, wqkv + (size_t)512 * 1024, cw);
        convert_f32_bf16<<<cw / (256 * 4), 256, 0, stream>>>(Wv, wqkv + (size_t)1024 * 1024, cw);
        convert_f32_bf16<<<cw / (256 * 4), 256, 0, stream>>>(Wo, wob, cw);
    }
    // 64x64 block tiles, 4 waves each; TDM double-buffered B staging.
    qkv_gemm<<<3072, 128, 0, stream>>>(xbf, wqkv, qb, kb, vt);
    mask_gemm<<<4096, 128, 0, stream>>>(xbf, mask);
    attn_kernel<<<1024, 128, 0, stream>>>(qb, kb, vt, mask, enhance, ob);
    out_gemm<<<2048, 128, 0, stream>>>(ob, wob, bo, out);
}
